// MultiHeadAttention_2619930050793
// MI455X (gfx1250) — compile-verified, matrix operand order checked by reading, not run on hardware
//
// CDNA5 (gfx1250) fused multi-head attention forward.
// bf16 WMMA (v_wmma_f32_16x16x32_bf16) everywhere; 302KB-LDS fused softmax;
// TDM (tensor_load_to_lds + s_wait_tensorcnt) for bf16 tile staging with
// double-buffered LDS; non-temporal stores for the 537MB attn tensor.

#include <hip/hip_runtime.h>
#include <hip/hip_bf16.h>

#define DEV __device__ __forceinline__

typedef __attribute__((ext_vector_type(16))) __bf16       v16bf;
typedef __attribute__((ext_vector_type(8)))  float        v8f;
typedef __attribute__((ext_vector_type(8)))  unsigned int v8u;

constexpr int NHEAD = 16, DMODEL = 1024, DK = 64, BATCH = 2, SEQ = 2048;
constexpr int MROWS = BATCH * SEQ;           // 4096 total token rows

// ---- TDM availability (device pass only; host pass parses fallback) ----
#if __has_builtin(__builtin_amdgcn_tensor_load_to_lds)
#define USE_TDM 1
#if __has_include(<hip/amd_detail/amd_gfx1250_TDM.h>)
#define TDM_6ARG 1               // amdgpu-toolchain (therock) 6-arg form
#else
#define TDM_6ARG 0               // ROCm 7.2 5-arg form
#endif
#else
#define USE_TDM 0
#endif

// ---------------- helpers ----------------

DEV unsigned short f2bf(float f) {           // round-to-nearest-even f32->bf16
  unsigned int u = __float_as_uint(f);
  u += 0x7FFFu + ((u >> 16) & 1u);
  return (unsigned short)(u >> 16);
}

DEV unsigned int pack2(unsigned short lo, unsigned short hi) {
  return (unsigned int)lo | ((unsigned int)hi << 16);
}

DEV v8f wmma_bf16(v16bf a, v16bf b, v8f c) {
  // (neg_a, A, neg_b, B, c_mod, C, reuse_a, reuse_b)
  return __builtin_amdgcn_wmma_f32_16x16x32_bf16(false, a, false, b, (short)0, c,
                                                 false, false);
}

#if USE_TDM
typedef __attribute__((ext_vector_type(4))) unsigned int v4u;
typedef __attribute__((ext_vector_type(4))) int          v4i;
typedef __attribute__((ext_vector_type(8))) int          v8i;

// 2-D bf16 tile DMA: global (row stride stride0 elems) -> LDS, with 1-DWORD
// row padding every (1<<padCode)*8 bytes (ISA 8.4 pad_interval/pad_amount).
// Descriptor per ISA 8.3/8.4: count=1, type=2, data_size=2B.
DEV void tdm_load_2d_bf16(void* ldsPtr, const unsigned short* gptr,
                          unsigned td0, unsigned td1, unsigned stride0,
                          unsigned tile0, unsigned tile1, unsigned padCode) {
  unsigned long long ga = (unsigned long long)(size_t)gptr;
  v4u g0 = { 1u,                                          // count=1 (valid)
             (unsigned)(size_t)ldsPtr,                    // lds_addr
             (unsigned)ga,                                // global_addr[31:0]
             ((unsigned)(ga >> 32) & 0x01FFFFFFu) | 0x80000000u }; // type=2
  v8i g1 = { (int)((1u << 16) | (1u << 20) | (padCode << 22)), // 2B,pad_en,interval
             (int)(td0 << 16),                            // tensor_dim0[15:0]
             (int)(td1 << 16),                            // tensor_dim1[15:0]
             (int)(tile0 << 16),                          // tile_dim0
             (int)tile1,                                  // tile_dim1
             (int)stride0,                                // tensor_dim0_stride
             0, 0 };
  v4i gz = {0, 0, 0, 0};
#if TDM_6ARG
  v8i gz8 = {0, 0, 0, 0, 0, 0, 0, 0};
  __builtin_amdgcn_tensor_load_to_lds(g0, g1, gz, gz, gz8, 0);
#else
  __builtin_amdgcn_tensor_load_to_lds(g0, g1, gz, gz, 0);
#endif
}
#endif  // USE_TDM

// 16x32 bf16 A or B fragment from a row-major LDS ushort matrix (ISA 7.12.2):
// lane&15 selects M (A) / N (B) row; K pairs per VGPR with the 8*half swizzle.
DEV v16bf frag_ab_lds(const unsigned short* base, int ld, int r0, int k0, int lane) {
  const int hlf = (lane >> 4) & 1;
  const unsigned short* row = base + (r0 + (lane & 15)) * ld;
  const int kb = k0 + 8 * hlf;
  v8u u;
#pragma unroll
  for (int r = 0; r < 4; ++r) {
    u[r]     = *(const unsigned int*)(row + kb + 2 * r);
    u[r + 4] = *(const unsigned int*)(row + kb + 16 + 2 * r);
  }
  return __builtin_bit_cast(v16bf, u);
}

// Same A-fragment gather from f32 LDS, converted to bf16 on the fly.
DEV v16bf frag_a_f32(const float* base, int ld, int r0, int k0, int lane) {
  const int hlf = (lane >> 4) & 1;
  const float* row = base + (size_t)(r0 + (lane & 15)) * ld;
  const int kb = k0 + 8 * hlf;
  v8u u;
#pragma unroll
  for (int r = 0; r < 4; ++r) {
    u[r]     = pack2(f2bf(row[kb + 2 * r]),      f2bf(row[kb + 2 * r + 1]));
    u[r + 4] = pack2(f2bf(row[kb + 16 + 2 * r]), f2bf(row[kb + 16 + 2 * r + 1]));
  }
  return __builtin_bit_cast(v16bf, u);
}

// ---------------- kernel 1: input projections ----------------
// C[4096x1024] = A(f32)[4096x1024] x W(f32)[1024x1024]; result * outScale
// stored bf16 into per-head layout dst[b][h][s][64]. Double-buffered LDS,
// one barrier per K-iteration (next-tile staging overlaps current WMMAs).
__global__ __launch_bounds__(256)
void proj_gemm_kernel(const float* __restrict__ A, const float* __restrict__ W,
                      unsigned short* __restrict__ dst, float outScale) {
  constexpr int BM = 128, BN = 128, BK = 32, LDT = BK + 2, TSZ = BM * LDT;
  __shared__ unsigned short As[2 * TSZ];    // [m][k] bf16, double buffered
  __shared__ unsigned short Bs[2 * TSZ];    // [n][k] bf16 (transposed on store)
  const int tid = threadIdx.x, lane = tid & 31, wave = tid >> 5;
  const int wm = wave & 1, wn = wave >> 1;  // 2 x 4 wave grid
  const int bm = blockIdx.y, bn = blockIdx.x;

  auto stageA = [&](unsigned short* as, int k0) {
#pragma unroll
    for (int i = 0; i < 4; ++i) {
      int e = tid + i * 256;                // 1024 float4 (128x32)
      int r = e >> 3, c4 = (e & 7) << 2;
      float4 f = *(const float4*)(A + (size_t)(bm * BM + r) * DMODEL + k0 + c4);
      unsigned short* d = &as[r * LDT + c4];
      d[0] = f2bf(f.x); d[1] = f2bf(f.y); d[2] = f2bf(f.z); d[3] = f2bf(f.w);
    }
  };
  auto stageB = [&](unsigned short* bs, int k0) {
#pragma unroll
    for (int i = 0; i < 4; ++i) {
      int e = tid + i * 256;                // 1024 float4 (32x128)
      int r = e >> 5, c4 = (e & 31) << 2;
      float4 f = *(const float4*)(W + (size_t)(k0 + r) * DMODEL + bn * BN + c4);
      bs[(c4 + 0) * LDT + r] = f2bf(f.x);
      bs[(c4 + 1) * LDT + r] = f2bf(f.y);
      bs[(c4 + 2) * LDT + r] = f2bf(f.z);
      bs[(c4 + 3) * LDT + r] = f2bf(f.w);
    }
  };

  const v8f vzero = {};
  v8f acc[4][2];
#pragma unroll
  for (int mi = 0; mi < 4; ++mi)
#pragma unroll
    for (int ni = 0; ni < 2; ++ni) acc[mi][ni] = vzero;

  constexpr int NIT = DMODEL / BK;          // 32
  stageA(As, 0);
  stageB(Bs, 0);
  for (int it = 0; it < NIT; ++it) {
    __syncthreads();
    const int cur = it & 1, nxt = cur ^ 1;
    if (it + 1 < NIT) {                     // overlap staging with WMMAs
      stageA(As + nxt * TSZ, (it + 1) * BK);
      stageB(Bs + nxt * TSZ, (it + 1) * BK);
    }
    const unsigned short* a = As + cur * TSZ;
    const unsigned short* b = Bs + cur * TSZ;
    v16bf bfr[2];
#pragma unroll
    for (int ni = 0; ni < 2; ++ni)
      bfr[ni] = frag_ab_lds(b, LDT, wn * 32 + ni * 16, 0, lane);
#pragma unroll
    for (int mi = 0; mi < 4; ++mi) {
      v16bf afr = frag_ab_lds(a, LDT, wm * 64 + mi * 16, 0, lane);
#pragma unroll
      for (int ni = 0; ni < 2; ++ni)
        acc[mi][ni] = wmma_bf16(afr, bfr[ni], acc[mi][ni]);
    }
  }

  // epilogue: scatter bf16 into [b][h][s][d]
  const int hlf = lane >> 4, n = lane & 15;
#pragma unroll
  for (int mi = 0; mi < 4; ++mi)
#pragma unroll
    for (int ni = 0; ni < 2; ++ni)
#pragma unroll
      for (int v = 0; v < 8; ++v) {
        int row = bm * BM + wm * 64 + mi * 16 + 8 * hlf + v;  // b*S + s
        int col = bn * BN + wn * 32 + ni * 16 + n;            // h*64 + d
        int bb = row >> 11, s = row & (SEQ - 1);
        int h = col >> 6, d = col & (DK - 1);
        dst[((size_t)(bb * NHEAD + h) * SEQ + s) * DK + d] =
            f2bf(acc[mi][ni][v] * outScale);
      }
}

// ---------------- kernel 2: fused attention per (b,h, 32-row block) ---------
// LDS plan (dynamic, 308,992 B <= 320KB WGP LDS):
//   Ss  : f32  [32][2048]   score/prob panel      262144 B
//   Qs  : bf16 [32][66]     staged Q tile           4224 B
//   Ks  : bf16 2x[128][66]  K panels (TDM dbl-buf) 33792 B
//   Vt  : bf16 2x[64][34]   V^T chunks (dbl-buf)    8704 B
//   rowScale : f32 [32]                              128 B
constexpr unsigned A_SS  = 0;
constexpr unsigned A_QS  = 262144;
constexpr unsigned A_KS0 = A_QS + 32 * 66 * 2;     // 266368
constexpr unsigned KS_BYTES = 128 * 66 * 2;        // 16896
constexpr unsigned A_VT0 = A_KS0 + 2 * KS_BYTES;   // 300160
constexpr unsigned VT_BYTES = 64 * 34 * 2;         // 4352
constexpr unsigned A_RS  = A_VT0 + 2 * VT_BYTES;   // 308864
constexpr unsigned ATTN_LDS_BYTES = A_RS + 32 * 4; // 308992

__global__ __launch_bounds__(256)
void attn_kernel(const unsigned short* __restrict__ Qh,
                 const unsigned short* __restrict__ Kh,
                 const unsigned short* __restrict__ Vh,
                 float* __restrict__ attnOut,
                 unsigned short* __restrict__ Ctx) {
  extern __shared__ char smem[];
  constexpr int ROWS = 32, KBN = 128, LDQ = DK + 2, LDV = 34;
  float*          Ss = (float*)(smem + A_SS);
  unsigned short* Qs = (unsigned short*)(smem + A_QS);
  float*    rowScale = (float*)(smem + A_RS);
  // NOTE: no arrays of LDS pointers (ld.lld can't relocate addrspacecast in
  // static initializers) — select buffers with index arithmetic instead.
  auto ksPtr = [&](int i) -> unsigned short* {
    return (unsigned short*)(smem + A_KS0 + (unsigned)i * KS_BYTES);
  };
  auto vtPtr = [&](int i) -> unsigned short* {
    return (unsigned short*)(smem + A_VT0 + (unsigned)i * VT_BYTES);
  };

  const int tid = threadIdx.x, lane = tid & 31, wave = tid >> 5;
  const int rb = blockIdx.x;                 // 64 row blocks of 32 queries
  const int bh = blockIdx.y;                 // b*16 + h
  const unsigned short* Qp = Qh + (size_t)bh * SEQ * DK + (size_t)rb * ROWS * DK;
  const unsigned short* Kp = Kh + (size_t)bh * SEQ * DK;
  const unsigned short* Vp = Vh + (size_t)bh * SEQ * DK;
  float* attnP = attnOut + (size_t)bh * SEQ * SEQ + (size_t)rb * ROWS * SEQ;

  // stage Q block (32x64 bf16, already pre-scaled by 1/sqrt(DK))
#pragma unroll
  for (int i = 0; i < 4; ++i) {
    int e = tid + i * 256;
    int r = e >> 5, c2 = e & 31;
    *(unsigned int*)&Qs[r * LDQ + (c2 << 1)] =
        *(const unsigned int*)&Qp[r * DK + (c2 << 1)];
  }

  auto stageK = [&](unsigned short* ks, int kb) {   // manual fallback
#pragma unroll
    for (int i = 0; i < 16; ++i) {
      int e = tid + i * 256;
      int r = e >> 5, c2 = e & 31;
      *(unsigned int*)&ks[r * LDQ + (c2 << 1)] =
          *(const unsigned int*)&Kp[(size_t)(kb + r) * DK + (c2 << 1)];
    }
  };

  // ---- phase 1: scores S = Q K^T into LDS f32 panel ----
  const int wm = wave & 1, wn = wave >> 1;
  const v8f vzero = {};
  constexpr int NKB = SEQ / KBN;             // 16 key panels
#if USE_TDM
  // 128x64 bf16 panel, 4B pad every 128B row -> 66-ushort LDS rows.
  if (wave == 0) tdm_load_2d_bf16(ksPtr(0), Kp, DK, SEQ, DK, DK, KBN, 4u);
#else
  stageK(ksPtr(0), 0);
#endif
  for (int it = 0; it < NKB; ++it) {
    const int kb = it * KBN;
    const bool hasNext = (it + 1 < NKB);
    __syncthreads();                          // prior compute on nxt buf done
#if USE_TDM
    if (wave == 0) {
      if (hasNext) {
        tdm_load_2d_bf16(ksPtr((it + 1) & 1), Kp + (size_t)(kb + KBN) * DK,
                         DK, SEQ, DK, DK, KBN, 4u);
        __builtin_amdgcn_s_wait_tensorcnt(1); // current panel landed
      } else {
        __builtin_amdgcn_s_wait_tensorcnt(0);
      }
    }
#else
    if (hasNext) stageK(ksPtr((it + 1) & 1), kb + KBN);
#endif
    __syncthreads();
    const unsigned short* Ks = ksPtr(it & 1);
    v8f acc[2] = {vzero, vzero};
#pragma unroll
    for (int ks = 0; ks < 2; ++ks) {          // DK=64 -> two K=32 steps
      v16bf a = frag_ab_lds(Qs, LDQ, wm * 16, ks * 32, lane);
#pragma unroll
      for (int ni = 0; ni < 2; ++ni) {
        v16bf b = frag_ab_lds(Ks, LDQ, wn * 32 + ni * 16, ks * 32, lane);
        acc[ni] = wmma_bf16(a, b, acc[ni]);
      }
    }
    const int hlf = lane >> 4, n = lane & 15;
#pragma unroll
    for (int ni = 0; ni < 2; ++ni)
#pragma unroll
      for (int v = 0; v < 8; ++v)
        Ss[(size_t)(wm * 16 + 8 * hlf + v) * SEQ + kb + wn * 32 + ni * 16 + n] =
            acc[ni][v];
  }
  __syncthreads();

  // ---- phase 2: softmax over each 2048-wide row (8 threads per row) ----
  {
    const int r = tid >> 3, sub = tid & 7;
    float* srow = Ss + (size_t)r * SEQ;
    float mx = -3.4e38f;
    for (int j = sub; j < SEQ; j += 8) mx = fmaxf(mx, srow[j]);
#pragma unroll
    for (int o = 1; o < 8; o <<= 1) mx = fmaxf(mx, __shfl_xor(mx, o, 8));
    float sum = 0.f;
    for (int j = sub; j < SEQ; j += 8) {
      float ev = __expf(srow[j] - mx);
      srow[j] = ev;
      sum += ev;
    }
#pragma unroll
    for (int o = 1; o < 8; o <<= 1) sum += __shfl_xor(sum, o, 8);
    if (sub == 0) rowScale[r] = 1.0f / sum;
  }
  __syncthreads();

  // ---- phase 3: normalize in LDS + stream attn to HBM (non-temporal) ----
  for (int i = 0; i < (ROWS * SEQ) / 256; ++i) {
    int e = tid + i * 256;
    int rr = e >> 11, c = e & (SEQ - 1);
    float p = Ss[(size_t)rr * SEQ + c] * rowScale[rr];
    Ss[(size_t)rr * SEQ + c] = p;
    __builtin_nontemporal_store(p, attnP + (size_t)rr * SEQ + c);
  }

  // ---- phase 4: O = P V (P from LDS f32, cvt bf16 in-register) ----
  auto stageV = [&](unsigned short* vt, int sb) {  // 32x64 -> Vt[d][s]
#pragma unroll
    for (int i = 0; i < 4; ++i) {
      int e = tid + i * 256;
      int sl = e >> 5, c2 = e & 31;
      unsigned int val =
          *(const unsigned int*)&Vp[(size_t)(sb + sl) * DK + (c2 << 1)];
      vt[(2 * c2) * LDV + sl]     = (unsigned short)(val & 0xFFFFu);
      vt[(2 * c2 + 1) * LDV + sl] = (unsigned short)(val >> 16);
    }
  };
  stageV(vtPtr(0), 0);
  v8f oacc = vzero;                           // one 16x16 tile per wave (2x4)
  constexpr int NSB = SEQ / 32;               // 64 chunks
  for (int it = 0; it < NSB; ++it) {
    __syncthreads();
    if (it + 1 < NSB) stageV(vtPtr((it + 1) & 1), (it + 1) * 32);
    v16bf a = frag_a_f32(Ss, SEQ, wm * 16, it * 32, lane);
    v16bf b = frag_ab_lds(vtPtr(it & 1), LDV, wn * 16, 0, lane);
    oacc = wmma_bf16(a, b, oacc);
  }

  // context store bf16 into [b][s][h*64+d] for the final projection
  const int hlf = lane >> 4, n = lane & 15;
  const int bb = bh >> 4, h = bh & (NHEAD - 1);
#pragma unroll
  for (int v = 0; v < 8; ++v) {
    int s = rb * ROWS + wm * 16 + 8 * hlf + v;
    int d = wn * 16 + n;
    Ctx[((size_t)bb * SEQ + s) * DMODEL + h * DK + d] = f2bf(oacc[v]);
  }
}

// ---------------- kernel 3: output projection ----------------
// out(f32)[4096x1024] = Ctx(bf16)[4096x1024] x w_fc(f32)[1024x1024]
// A tile is a pure bf16 strided copy -> staged by TDM; W tile manual.
__global__ __launch_bounds__(256)
void fc_gemm_kernel(const unsigned short* __restrict__ A,
                    const float* __restrict__ W, float* __restrict__ out) {
  constexpr int BM = 128, BN = 128, BK = 32, LDT = BK + 2, TSZ = BM * LDT;
  __shared__ unsigned short As[2 * TSZ];
  __shared__ unsigned short Bs[2 * TSZ];
  const int tid = threadIdx.x, lane = tid & 31, wave = tid >> 5;
  const int wm = wave & 1, wn = wave >> 1;
  const int bm = blockIdx.y, bn = blockIdx.x;

  auto stageA = [&](unsigned short* as, int k0) {  // manual fallback
#pragma unroll
    for (int i = 0; i < 8; ++i) {
      int e = tid + i * 256;
      int r = e >> 4, c2 = e & 15;
      *(unsigned int*)&as[r * LDT + (c2 << 1)] =
          *(const unsigned int*)(A + (size_t)(bm * BM + r) * DMODEL + k0 +
                                 (c2 << 1));
    }
  };
  auto stageB = [&](unsigned short* bs, int k0) {
#pragma unroll
    for (int i = 0; i < 4; ++i) {
      int e = tid + i * 256;
      int r = e >> 5, c4 = (e & 31) << 2;
      float4 f = *(const float4*)(W + (size_t)(k0 + r) * DMODEL + bn * BN + c4);
      bs[(c4 + 0) * LDT + r] = f2bf(f.x);
      bs[(c4 + 1) * LDT + r] = f2bf(f.y);
      bs[(c4 + 2) * LDT + r] = f2bf(f.z);
      bs[(c4 + 3) * LDT + r] = f2bf(f.w);
    }
  };

  const v8f vzero = {};
  v8f acc[4][2];
#pragma unroll
  for (int mi = 0; mi < 4; ++mi)
#pragma unroll
    for (int ni = 0; ni < 2; ++ni) acc[mi][ni] = vzero;

  constexpr int NIT = DMODEL / BK;          // 32
  const unsigned short* Abase = A + (size_t)bm * BM * DMODEL;
#if USE_TDM
  // 128x32 bf16 tile, 4B pad every 64B row -> 34-ushort LDS rows.
  if (wave == 0)
    tdm_load_2d_bf16(As, Abase, DMODEL, BM, DMODEL, BK, BM, 3u);
#else
  stageA(As, 0);
#endif
  stageB(Bs, 0);
  for (int it = 0; it < NIT; ++it) {
    __syncthreads();
    const int cur = it & 1, nxt = cur ^ 1;
    const bool hasNext = (it + 1 < NIT);
    if (hasNext) stageB(Bs + nxt * TSZ, (it + 1) * BK);
#if USE_TDM
    if (wave == 0) {
      if (hasNext) {
        tdm_load_2d_bf16(As + nxt * TSZ, Abase + (it + 1) * BK,
                         DMODEL, BM, DMODEL, BK, BM, 3u);
        __builtin_amdgcn_s_wait_tensorcnt(1);
      } else {
        __builtin_amdgcn_s_wait_tensorcnt(0);
      }
    }
    __syncthreads();
#else
    if (hasNext) stageA(As + nxt * TSZ, (it + 1) * BK);
#endif
    const unsigned short* a = As + cur * TSZ;
    const unsigned short* b = Bs + cur * TSZ;
    v16bf bfr[2];
#pragma unroll
    for (int ni = 0; ni < 2; ++ni)
      bfr[ni] = frag_ab_lds(b, LDT, wn * 32 + ni * 16, 0, lane);
#pragma unroll
    for (int mi = 0; mi < 4; ++mi) {
      v16bf afr = frag_ab_lds(a, LDT, wm * 64 + mi * 16, 0, lane);
#pragma unroll
      for (int ni = 0; ni < 2; ++ni)
        acc[mi][ni] = wmma_bf16(afr, bfr[ni], acc[mi][ni]);
    }
  }

  const int hlf = lane >> 4, n = lane & 15;
#pragma unroll
  for (int mi = 0; mi < 4; ++mi)
#pragma unroll
    for (int ni = 0; ni < 2; ++ni)
#pragma unroll
      for (int v = 0; v < 8; ++v) {
        int row = bm * BM + wm * 64 + mi * 16 + 8 * hlf + v;
        int col = bn * BN + wn * 32 + ni * 16 + n;
        out[(size_t)row * DMODEL + col] = acc[mi][ni][v];
      }
}

// ---------------- launch ----------------
extern "C" void kernel_launch(void* const* d_in, const int* in_sizes, int n_in,
                              void* d_out, int out_size, void* d_ws, size_t ws_size,
                              hipStream_t stream) {
  (void)in_sizes; (void)n_in; (void)out_size; (void)ws_size;
  const float* q    = (const float*)d_in[0];
  const float* k    = (const float*)d_in[1];
  const float* v    = (const float*)d_in[2];
  const float* w_qs = (const float*)d_in[3];
  const float* w_ks = (const float*)d_in[4];
  const float* w_vs = (const float*)d_in[5];
  const float* w_fc = (const float*)d_in[6];

  float* out  = (float*)d_out;                       // [B,S,1024] f32
  float* attn = out + (size_t)MROWS * DMODEL;        // [B,H,S,S]  f32

  // workspace: Qh | Kh | Vh | Ctx, all bf16 (8 MB each, 32 MB total)
  const size_t per = (size_t)BATCH * NHEAD * SEQ * DK;
  unsigned short* Qh  = (unsigned short*)d_ws;
  unsigned short* Kh  = Qh + per;
  unsigned short* Vh  = Kh + per;
  unsigned short* Ctx = Vh + per;

  dim3 gg(DMODEL / 128, MROWS / 128);                // 8 x 32 blocks
  proj_gemm_kernel<<<gg, 256, 0, stream>>>(q, w_qs, Qh, 0.125f);  // 1/sqrt(DK)
  proj_gemm_kernel<<<gg, 256, 0, stream>>>(k, w_ks, Kh, 1.0f);
  proj_gemm_kernel<<<gg, 256, 0, stream>>>(v, w_vs, Vh, 1.0f);

  attn_kernel<<<dim3(SEQ / 32, BATCH * NHEAD), 256, ATTN_LDS_BYTES, stream>>>(
      Qh, Kh, Vh, attn, Ctx);

  fc_gemm_kernel<<<gg, 256, 0, stream>>>(Ctx, w_fc, out);
}
